// CPBTransBlock_82085414961219
// MI455X (gfx1250) — matrix operand order repaired and checked by reading, not compiled, not run on hardware
//
#include <hip/hip_runtime.h>
#include <hip/hip_bf16.h>
#include <math.h>

// ---------------------------------------------------------------------------
// CDNA5 (gfx1250) types and WMMA helpers
// ---------------------------------------------------------------------------
typedef __bf16 bf16_t;
typedef __attribute__((ext_vector_type(16))) __bf16 v16bf;
typedef __attribute__((ext_vector_type(8)))  float  v8f;

__device__ __forceinline__ v8f wmma_bf16(v16bf a, v16bf b, v8f c) {
  // D = A(16x32) * B(32x16) + C, fp32 accumulate
  return __builtin_amdgcn_wmma_f32_16x16x32_bf16(
      /*neg_a=*/false, a, /*neg_b=*/false, b,
      /*c_mod=*/(short)0, c, /*reuse_a=*/false, /*reuse_b=*/false);
}

// ---- async global->LDS copies (ISA ch. 10.7/15.18, tracked by ASYNCcnt) ----
__device__ __forceinline__ void async_ld_b32(void* lds, const void* g) {
  unsigned l = (unsigned)(unsigned long long)lds;  // low 32 bits = LDS offset
  asm volatile("global_load_async_to_lds_b32 %0, %1, off"
               :: "v"(l), "v"(g) : "memory");
}
__device__ __forceinline__ void async_ld_b128(void* lds, const void* g) {
  unsigned l = (unsigned)(unsigned long long)lds;
  asm volatile("global_load_async_to_lds_b128 %0, %1, off"
               :: "v"(l), "v"(g) : "memory");
}

// ---------------------------------------------------------------------------
// Fragment builders (contiguous k = 16h + s, same order in A and B).
// A (16x32 MxK): lanes0-15 row M=lane hold K=0..15;
//                lanes16-31 K=16..31.
// B (32x16 KxN): lanes0-15 col N=lane hold K=0..15; lanes16-31 K=16..31.
// ---------------------------------------------------------------------------
// A, element (m,k) at base[m*ldm + k], bf16 source
__device__ __forceinline__ v16bf fragA_lds(const bf16_t* base, int ldm, int lane) {
  int m = lane & 15;
  int kb = (lane & 16) ? 16 : 0;
  const bf16_t* row = base + m * ldm;
  v16bf a;
#pragma unroll
  for (int r = 0; r < 4; ++r) {
    a[2*r]       = row[kb + 2*r];
    a[2*r+1]     = row[kb + 2*r + 1];
    a[8 + 2*r]   = row[8 + kb + 2*r];
    a[8 + 2*r+1] = row[8 + kb + 2*r + 1];
  }
  return a;
}
// A, element (m,k) at base[m*ldm + k], fp32 source
__device__ __forceinline__ v16bf fragA_lds_f32(const float* base, int ldm, int lane) {
  int m = lane & 15;
  int kb = (lane & 16) ? 16 : 0;
  const float* row = base + m * ldm;
  v16bf a;
#pragma unroll
  for (int r = 0; r < 4; ++r) {
    a[2*r]       = (bf16_t)row[kb + 2*r];
    a[2*r+1]     = (bf16_t)row[kb + 2*r + 1];
    a[8 + 2*r]   = (bf16_t)row[8 + kb + 2*r];
    a[8 + 2*r+1] = (bf16_t)row[8 + kb + 2*r + 1];
  }
  return a;
}
// A from transposed storage: element (m,k) at base[k*ldk + m], bf16
__device__ __forceinline__ v16bf fragAt_lds(const bf16_t* base, int ldk, int lane) {
  int m = lane & 15;
  int kb = (lane & 16) ? 16 : 0;
  v16bf a;
#pragma unroll
  for (int r = 0; r < 4; ++r) {
    a[2*r]       = base[(kb + 2*r) * ldk + m];
    a[2*r+1]     = base[(kb + 2*r + 1) * ldk + m];
    a[8 + 2*r]   = base[(8 + kb + 2*r) * ldk + m];
    a[8 + 2*r+1] = base[(8 + kb + 2*r + 1) * ldk + m];
  }
  return a;
}
// A from transposed storage, fp32 source
__device__ __forceinline__ v16bf fragAt_lds_f32(const float* base, int ldk, int lane) {
  int m = lane & 15;
  int kb = (lane & 16) ? 16 : 0;
  v16bf a;
#pragma unroll
  for (int r = 0; r < 4; ++r) {
    a[2*r]       = (bf16_t)base[(kb + 2*r) * ldk + m];
    a[2*r+1]     = (bf16_t)base[(kb + 2*r + 1) * ldk + m];
    a[8 + 2*r]   = (bf16_t)base[(8 + kb + 2*r) * ldk + m];
    a[8 + 2*r+1] = (bf16_t)base[(8 + kb + 2*r + 1) * ldk + m];
  }
  return a;
}
// B, element (k,n) at base[k*ldk + n], fp32 source
__device__ __forceinline__ v16bf fragB_lds_f32(const float* base, int ldk, int lane) {
  int n = lane & 15;
  int kb = (lane & 16) ? 16 : 0;
  v16bf b;
#pragma unroll
  for (int r = 0; r < 8; ++r) {
    b[2*r]   = (bf16_t)base[(kb + 2*r) * ldk + n];
    b[2*r+1] = (bf16_t)base[(kb + 2*r + 1) * ldk + n];
  }
  return b;
}
// B from transposed storage: element (k,n) at base[n*ldn + k], bf16
__device__ __forceinline__ v16bf fragBt_lds(const bf16_t* base, int ldn, int lane) {
  int n = lane & 15;
  int kb = (lane & 16) ? 16 : 0;
  const bf16_t* col = base + n * ldn;
  v16bf b;
#pragma unroll
  for (int r = 0; r < 8; ++r) {
    b[2*r]   = col[kb + 2*r];
    b[2*r+1] = col[kb + 2*r + 1];
  }
  return b;
}
// B from transposed storage, fp32 source
__device__ __forceinline__ v16bf fragBt_lds_f32(const float* base, int ldn, int lane) {
  int n = lane & 15;
  int kb = (lane & 16) ? 16 : 0;
  const float* col = base + n * ldn;
  v16bf b;
#pragma unroll
  for (int r = 0; r < 8; ++r) {
    b[2*r]   = (bf16_t)col[kb + 2*r];
    b[2*r+1] = (bf16_t)col[kb + 2*r + 1];
  }
  return b;
}

__device__ __forceinline__ float gelu_exact(float v) {
  return 0.5f * v * (1.0f + erff(v * 0.70710678118654752f));
}

// ---------------------------------------------------------------------------
// Generic token GEMM:  Out[n_tok, o] = sum_c A(tok, c) * W[o*Cin + c] + bias[o]
//  A_CN : A stored channel-major A[c*NA + tok] (else token-major A[tok*Cin+c])
//  OUT_CN: Out channel-major Out[o*NOut + tok] (else Out[tok*Cout+o])
//  GATHER: token rows indirected through gidx (compact output rows)
//  ACT=1: exact GELU
// 64x64 tile / 128 threads; staging via double-buffered async global->LDS
// (fp32 in LDS, bf16 conversion folded into fragment gather).
// LDS layouts: A_CN  -> sA[k][64 tokens]  (async rows contiguous over tokens)
//              !A_CN -> sA[m][32 k]       (async rows contiguous over k)
//              sB[n][32 k]                (async rows contiguous over k)
// ---------------------------------------------------------------------------
template <bool A_CN, bool OUT_CN, bool GATHER, int ACT>
__global__ __launch_bounds__(128)
void gemm_tok(const float* __restrict__ A, const float* __restrict__ W,
              const float* __restrict__ bias, float* __restrict__ Out,
              const int* __restrict__ gidx,
              int NA, int NOut, int Cin, int Cout) {
  __shared__ alignas(16) float sA[2][2048];
  __shared__ alignas(16) float sB[2][2048];

  const int tid  = threadIdx.x;
  const int lane = tid & 31;
  const int wv   = tid >> 5;
  const int gm0  = blockIdx.x * 64;
  const int gn0  = blockIdx.y * 64;
  const int wm   = (wv & 1) * 32;
  const int wn   = (wv >> 1) * 32;
  const int nk   = Cin >> 5;

  auto issue = [&](int buf, int k0) {
    if (A_CN) {
      if (GATHER) {
#pragma unroll
        for (int i = 0; i < 16; ++i) {      // per-lane scatter-gather, b32
          int e = tid + i * 128;
          int k = e >> 6, m = e & 63;
          int tok = gidx[gm0 + m];
          async_ld_b32(&sA[buf][k * 64 + m], &A[(size_t)(k0 + k) * NA + tok]);
        }
      } else {
#pragma unroll
        for (int i = 0; i < 4; ++i) {       // contiguous over tokens, b128
          int ch = tid + i * 128;
          int k = ch >> 4, cm = ch & 15;
          async_ld_b128(&sA[buf][k * 64 + cm * 4],
                        &A[(size_t)(k0 + k) * NA + gm0 + cm * 4]);
        }
      }
    } else {
#pragma unroll
      for (int i = 0; i < 4; ++i) {         // contiguous over k, b128
        int ch = tid + i * 128;
        int m = ch >> 3, cm = ch & 7;
        async_ld_b128(&sA[buf][m * 32 + cm * 4],
                      &A[(size_t)(gm0 + m) * Cin + k0 + cm * 4]);
      }
    }
#pragma unroll
    for (int i = 0; i < 4; ++i) {           // W rows contiguous over k, b128
      int ch = tid + i * 128;
      int n = ch >> 3, cm = ch & 7;
      async_ld_b128(&sB[buf][n * 32 + cm * 4],
                    &W[(size_t)(gn0 + n) * Cin + k0 + cm * 4]);
    }
  };
  constexpr int ISS = (A_CN && GATHER) ? 20 : 8;  // async issues per tile

  v8f acc00 = {}, acc01 = {}, acc10 = {}, acc11 = {};

  issue(0, 0);
  for (int kt = 0; kt < nk; ++kt) {
    const int cur = kt & 1;
    if (kt + 1 < nk) {
      issue(cur ^ 1, (kt + 1) << 5);
      asm volatile("s_wait_asynccnt %0" :: "i"(ISS) : "memory");
    } else {
      asm volatile("s_wait_asynccnt 0" ::: "memory");
    }
    __syncthreads();

    v16bf a0, a1;
    if (A_CN) {
      a0 = fragAt_lds_f32(&sA[cur][wm + 0],  64, lane);
      a1 = fragAt_lds_f32(&sA[cur][wm + 16], 64, lane);
    } else {
      a0 = fragA_lds_f32(&sA[cur][(wm + 0)  * 32], 32, lane);
      a1 = fragA_lds_f32(&sA[cur][(wm + 16) * 32], 32, lane);
    }
    v16bf b0 = fragBt_lds_f32(&sB[cur][(wn + 0)  * 32], 32, lane);
    v16bf b1 = fragBt_lds_f32(&sB[cur][(wn + 16) * 32], 32, lane);

    acc00 = wmma_bf16(a0, b0, acc00);
    acc01 = wmma_bf16(a0, b1, acc01);
    acc10 = wmma_bf16(a1, b0, acc10);
    acc11 = wmma_bf16(a1, b1, acc11);
    __syncthreads();
  }

  // epilogue: C/D layout — lanes0-15: M=r,N=lane; lanes16-31: M=8+r,N=lane-16
  const int nloc  = lane & 15;
  const int mbase = (lane >> 4) * 8;

  auto store_tile = [&](const v8f& acc, int ti, int tj) {
#pragma unroll
    for (int r = 0; r < 8; ++r) {
      int m = gm0 + wm + ti * 16 + mbase + r;   // output token row (compact)
      int n = gn0 + wn + tj * 16 + nloc;        // output feature
      float v = acc[r];
      if (bias) v += bias[n];
      if (ACT == 1) v = gelu_exact(v);
      if (OUT_CN) Out[(size_t)n * NOut + m] = v;
      else        Out[(size_t)m * Cout + n] = v;
    }
  };
  store_tile(acc00, 0, 0);
  store_tile(acc01, 0, 1);
  store_tile(acc10, 1, 0);
  store_tile(acc11, 1, 1);
}

// ---------------------------------------------------------------------------
// Flash attention over 512 gathered keys; one (64-query tile, head) per WG.
// Computes S^T = K@Q^T so the softmax reduction (over keys) is per-lane over
// the accumulator index + one cross-half shuffle; O^T = V^T @ P^T.
// q : (N, 256) row-major; kg/vg : (512, 256) row-major; o : (N, 256) row-major
// ---------------------------------------------------------------------------
__global__ __launch_bounds__(128)
void attn_kernel(const float* __restrict__ q, const float* __restrict__ kg,
                 const float* __restrict__ vg, float* __restrict__ o,
                 int N, int C) {
  __shared__ bf16_t sK[256 * 32];    // 16 KB  key chunk, key-major rows
  __shared__ bf16_t sV[256 * 32];    // 16 KB
  __shared__ bf16_t sQ[64 * 32];     // 4 KB   query-major rows
  __shared__ float  sP[4 * 32 * 16]; // 8 KB   per-wave P^T (32 keys x 16 q)

  const int tid  = threadIdx.x;
  const int lane = tid & 31;
  const int wv   = tid >> 5;
  const int head = blockIdx.y;
  const int gm0  = blockIdx.x * 64;
  const int wm   = wv * 16;

  for (int i = tid; i < 64 * 32; i += 128) {
    int r = i >> 5, d = i & 31;
    sQ[i] = (bf16_t)q[(size_t)(gm0 + r) * C + head * 32 + d];
  }
  __syncthreads();
  // B fragment: Q^T (32 dims x 16 queries) from query-major storage
  const v16bf qB = fragBt_lds(sQ + wm * 32, 32, lane);

  const float rs = 0.17677669529663687f;  // 1/sqrt(32)
  v8f o0 = {}, o1 = {};   // O^T tiles: (dims 0-15, q), (dims 16-31, q)
  float mrun = -INFINITY, lrun = 0.0f;   // per-lane stats for query nloc

  const int nloc  = lane & 15;
  const int mbase = (lane >> 4) * 8;
  float* myP = sP + wv * 32 * 16;

  for (int kh = 0; kh < 2; ++kh) {
    __syncthreads();
    for (int i = tid; i < 256 * 32; i += 128) {
      int kk = i >> 5, d = i & 31;
      int krow = kh * 256 + kk;
      sK[i] = (bf16_t)kg[(size_t)krow * C + head * 32 + d];
      sV[i] = (bf16_t)vg[(size_t)krow * C + head * 32 + d];
    }
    __syncthreads();

    for (int kt = 0; kt < 8; ++kt) {
      const int key0 = kt * 32;
      // S^T = K(16x32) @ Q^T(32x16); rows = keys, cols = queries
      v8f z0 = {}, z1 = {};
      v8f s0 = wmma_bf16(fragA_lds(sK + (key0 + 0)  * 32, 32, lane), qB, z0);
      v8f s1 = wmma_bf16(fragA_lds(sK + (key0 + 16) * 32, 32, lane), qB, z1);

      float tm = -INFINITY;
#pragma unroll
      for (int r = 0; r < 8; ++r) {
        s0[r] *= rs;
        s1[r] *= rs;
        tm = fmaxf(tm, fmaxf(s0[r], s1[r]));
      }
      tm = fmaxf(tm, __shfl_xor(tm, 16, 32));   // other half of key rows
      float mn   = fmaxf(mrun, tm);
      float corr = __expf(mrun - mn);
      float ps = 0.0f;
#pragma unroll
      for (int r = 0; r < 8; ++r) {
        float p0 = __expf(s0[r] - mn);
        float p1 = __expf(s1[r] - mn);
        ps += p0 + p1;
        myP[(mbase + r) * 16 + nloc]      = p0;  // keys key0+0..15
        myP[(16 + mbase + r) * 16 + nloc] = p1;  // keys key0+16..31
      }
      ps += __shfl_xor(ps, 16, 32);
      lrun = lrun * corr + ps;
      mrun = mn;
#pragma unroll
      for (int r = 0; r < 8; ++r) { o0[r] *= corr; o1[r] *= corr; }
      __syncthreads();
      // O^T += V^T(16x32 keys) @ P^T(32 keys x 16 queries)
      v16bf pB = fragB_lds_f32(myP, 16, lane);
      o0 = wmma_bf16(fragAt_lds(sV + key0 * 32 + 0,  32, lane), pB, o0);
      o1 = wmma_bf16(fragAt_lds(sV + key0 * 32 + 16, 32, lane), pB, o1);
      __syncthreads();
    }
  }

  // store: O^T element (dim = tile*16 + mbase + r, query = nloc)
  const float inv = 1.0f / lrun;
  const size_t qrow = (size_t)(gm0 + wm + nloc) * C + head * 32;
#pragma unroll
  for (int r = 0; r < 8; ++r) o[qrow + mbase + r] = o0[r] * inv;
#pragma unroll
  for (int r = 0; r < 8; ++r) o[qrow + 16 + mbase + r] = o1[r] * inv;
}

// ---------------------------------------------------------------------------
// Small kernels
// ---------------------------------------------------------------------------
__global__ void meanmax_kernel(const float* __restrict__ x, float* mean,
                               float* mx, int C, int N) {
  int n = blockIdx.x * blockDim.x + threadIdx.x;
  if (n >= N) return;
  float s = 0.0f, m = -INFINITY;
  for (int c = 0; c < C; ++c) {
    float v = x[(size_t)c * N + n];
    s += v;
    m = fmaxf(m, v);
  }
  mean[n] = s / (float)C;
  mx[n]   = m;
}

__global__ void spa_conv_kernel(const float* __restrict__ mean,
                                const float* __restrict__ mx,
                                const float* __restrict__ w,
                                float* __restrict__ sc, int D, int H, int W) {
  int n = blockIdx.x * blockDim.x + threadIdx.x;
  int HW = H * W, Nn = D * HW;
  if (n >= Nn) return;
  int d = n / HW, rem = n % HW, h = rem / W, wd = rem % W;
  float acc = 0.0f;
  for (int kd = 0; kd < 7; ++kd) {
    int dd = d + kd - 3;
    if (dd < 0 || dd >= D) continue;
    for (int kh = 0; kh < 7; ++kh) {
      int hh = h + kh - 3;
      if (hh < 0 || hh >= H) continue;
      for (int kw = 0; kw < 7; ++kw) {
        int ww = wd + kw - 3;
        if (ww < 0 || ww >= W) continue;
        int src = dd * HW + hh * W + ww;
        int wi  = kd * 49 + kh * 7 + kw;
        acc += w[wi] * mean[src] + w[343 + wi] * mx[src];
      }
    }
  }
  sc[n] = 1.0f / (1.0f + __expf(-acc));
}

__device__ __forceinline__ unsigned fkey(float f) {
  unsigned u = __float_as_uint(f);
  return (u & 0x80000000u) ? ~u : (u | 0x80000000u);
}

// Exact radix top-k selection (selected set is order-invariant downstream).
__global__ void topk_kernel(const float* __restrict__ scores,
                            int* __restrict__ idx, int N, int K) {
  __shared__ unsigned hist[256];
  __shared__ unsigned sprefix;
  __shared__ int sneed;
  __shared__ int scnt;
  int tid = threadIdx.x;
  if (tid == 0) { sprefix = 0u; sneed = K; scnt = 0; }
  __syncthreads();

  for (int round = 0; round < 3; ++round) {
    int shift = 24 - round * 8;
    unsigned pmask = (round == 0) ? 0u : (0xFFFFFFFFu << (32 - round * 8));
    if (tid < 256) hist[tid] = 0u;
    __syncthreads();
    unsigned prefix = sprefix;
    for (int n = tid; n < N; n += blockDim.x) {
      unsigned k = fkey(scores[n]);
      if ((k & pmask) == prefix) atomicAdd(&hist[(k >> shift) & 255u], 1u);
    }
    __syncthreads();
    if (tid == 0) {
      int need = sneed;
      unsigned cum = 0;
      int b = 255;
      for (; b > 0; --b) {
        if ((int)(cum + hist[b]) >= need) break;
        cum += hist[b];
      }
      sneed = need - (int)cum;
      sprefix = prefix | ((unsigned)b << shift);
    }
    __syncthreads();
  }

  unsigned thr = sprefix >> 8;
  for (int n = tid; n < N; n += blockDim.x) {
    if ((fkey(scores[n]) >> 8) > thr) {
      int p = atomicAdd(&scnt, 1);
      if (p < K) idx[p] = n;
    }
  }
  __syncthreads();
  if (tid == 0) {  // deterministic tie fill
    int p = scnt;
    for (int n = 0; n < N && p < K; ++n) {
      if ((fkey(scores[n]) >> 8) == thr) idx[p++] = n;
    }
  }
}

__global__ void gather_vg_kernel(const float* __restrict__ v,
                                 const int* __restrict__ idx,
                                 float* __restrict__ vg, int C, int N) {
  int e = blockIdx.x * blockDim.x + threadIdx.x;  // K*C
  int t = e >> 8, c = e & 255;
  vg[e] = v[(size_t)c * N + idx[t]];
}

__global__ void dwconv_kernel(const float* __restrict__ v,
                              const float* __restrict__ w,
                              const float* __restrict__ b,
                              float* __restrict__ out,
                              int C, int D, int H, int W) {
  int e = blockIdx.x * blockDim.x + threadIdx.x;
  int HW = H * W, N = D * HW;
  if (e >= C * N) return;
  int c = e / N, s = e % N;
  int d = s / HW, rem = s % HW, h = rem / W, wd = rem % W;
  float acc = b[c];
  for (int kd = 0; kd < 3; ++kd) {
    int dd = d + kd - 1;
    if (dd < 0 || dd >= D) continue;
    for (int kh = 0; kh < 3; ++kh) {
      int hh = h + kh - 1;
      if (hh < 0 || hh >= H) continue;
      for (int kw = 0; kw < 3; ++kw) {
        int ww = wd + kw - 1;
        if (ww < 0 || ww >= W) continue;
        acc += w[c * 27 + kd * 9 + kh * 3 + kw] *
               v[(size_t)c * N + dd * HW + hh * W + ww];
      }
    }
  }
  out[e] = acc;
}

__global__ void gap_gmp_kernel(const float* __restrict__ x, float* gap,
                               float* gmp, int N) {
  int c = blockIdx.x, tid = threadIdx.x;
  __shared__ float ssum[256];
  __shared__ float smax[256];
  float s = 0.0f, m = -INFINITY;
  for (int n = tid; n < N; n += 256) {
    float v = x[(size_t)c * N + n];
    s += v;
    m = fmaxf(m, v);
  }
  ssum[tid] = s; smax[tid] = m;
  __syncthreads();
  for (int off = 128; off > 0; off >>= 1) {
    if (tid < off) {
      ssum[tid] += ssum[tid + off];
      smax[tid] = fmaxf(smax[tid], smax[tid + off]);
    }
    __syncthreads();
  }
  if (tid == 0) { gap[c] = ssum[0] / (float)N; gmp[c] = smax[0]; }
}

__global__ void ca_fc_kernel(const float* __restrict__ gap,
                             const float* __restrict__ gmp,
                             const float* __restrict__ w1,
                             const float* __restrict__ w2,
                             float* __restrict__ ca, int C, int Cr) {
  __shared__ float ya[16];
  __shared__ float yb[16];
  int tid = threadIdx.x;  // 256 threads
  if (tid < Cr) {
    float sa = 0.0f, sb = 0.0f;
    for (int c = 0; c < C; ++c) {
      float w = w1[tid * C + c];
      sa += w * gap[c];
      sb += w * gmp[c];
    }
    ya[tid] = fmaxf(sa, 0.0f);
    yb[tid] = fmaxf(sb, 0.0f);
  }
  __syncthreads();
  float za = 0.0f, zb = 0.0f;
  for (int i = 0; i < Cr; ++i) {
    float w = w2[tid * Cr + i];
    za += w * ya[i];
    zb += w * yb[i];
  }
  ca[tid] = 1.0f / (1.0f + __expf(-(za + zb)));
}

__global__ void h_kernel(const float* __restrict__ x,
                         const float* __restrict__ ca,
                         const float* __restrict__ proj,
                         const float* __restrict__ pw,
                         float* __restrict__ h, int C, int N) {
  size_t e = (size_t)blockIdx.x * blockDim.x + threadIdx.x;
  if (e >= (size_t)C * N) return;
  int c = (int)(e / N);
  h[e] = x[e] * ca[c] + proj[e] + pw[e];
}

// LN over channel dim (h in CN layout): x1 = x + LN(h)*g + b
__global__ void ln1_kernel(const float* __restrict__ x,
                           const float* __restrict__ h,
                           const float* __restrict__ g,
                           const float* __restrict__ b,
                           float* __restrict__ x1, int C, int N) {
  int n = blockIdx.x * blockDim.x + threadIdx.x;
  if (n >= N) return;
  float m = 0.0f;
  for (int c = 0; c < C; ++c) m += h[(size_t)c * N + n];
  m /= (float)C;
  float v = 0.0f;
  for (int c = 0; c < C; ++c) {
    float d = h[(size_t)c * N + n] - m;
    v += d * d;
  }
  v /= (float)C;
  float rstd = rsqrtf(v + 1e-5f);
  for (int c = 0; c < C; ++c) {
    x1[(size_t)c * N + n] =
        x[(size_t)c * N + n] + (h[(size_t)c * N + n] - m) * rstd * g[c] + b[c];
  }
}

// out = x1 + LN(t)*g + b ; t is (N, C) row-major; out in CN layout
__global__ void ln2_kernel(const float* __restrict__ x1,
                           const float* __restrict__ t,
                           const float* __restrict__ g,
                           const float* __restrict__ b,
                           float* __restrict__ out, int C, int N) {
  int n = blockIdx.x * blockDim.x + threadIdx.x;
  if (n >= N) return;
  const float* row = t + (size_t)n * C;
  float m = 0.0f;
  for (int c = 0; c < C; ++c) m += row[c];
  m /= (float)C;
  float v = 0.0f;
  for (int c = 0; c < C; ++c) {
    float d = row[c] - m;
    v += d * d;
  }
  v /= (float)C;
  float rstd = rsqrtf(v + 1e-5f);
  for (int c = 0; c < C; ++c) {
    out[(size_t)c * N + n] =
        x1[(size_t)c * N + n] + (row[c] - m) * rstd * g[c] + b[c];
  }
}

// ---------------------------------------------------------------------------
// Host orchestration
// ---------------------------------------------------------------------------
extern "C" void kernel_launch(void* const* d_in, const int* in_sizes, int n_in,
                              void* d_out, int out_size, void* d_ws,
                              size_t ws_size, hipStream_t stream) {
  (void)in_sizes; (void)n_in; (void)out_size; (void)ws_size;

  const float* x      = (const float*)d_in[0];
  const float* x2     = (const float*)d_in[1];
  const float* sa_w   = (const float*)d_in[2];
  const float* kv_w   = (const float*)d_in[3];
  const float* kv_b   = (const float*)d_in[4];
  const float* q_w    = (const float*)d_in[5];
  const float* q_b    = (const float*)d_in[6];
  const float* proj_w = (const float*)d_in[7];
  const float* proj_b = (const float*)d_in[8];
  const float* dw_w   = (const float*)d_in[9];
  const float* dw_b   = (const float*)d_in[10];
  const float* pw_w   = (const float*)d_in[11];
  const float* pw_b   = (const float*)d_in[12];
  const float* ca_w1  = (const float*)d_in[13];
  const float* ca_w2  = (const float*)d_in[14];
  const float* ln1_g  = (const float*)d_in[15];
  const float* ln1_b  = (const float*)d_in[16];
  const float* ln2_g  = (const float*)d_in[17];
  const float* ln2_b  = (const float*)d_in[18];
  const float* mlp_w1 = (const float*)d_in[19];
  const float* mlp_b1 = (const float*)d_in[20];
  const float* mlp_w2 = (const float*)d_in[21];
  const float* mlp_b2 = (const float*)d_in[22];

  const int D = 16, H = 32, W = 32;
  const int N = D * H * W;        // 16384 tokens
  const int C = 256;
  const int K = 512;              // topk
  const int HEADS = 16;

  // workspace carve-out
  float* ws = (float*)d_ws;
  size_t off = 0;
  float* spa_mean = ws + off; off += N;
  float* spa_max  = ws + off; off += N;
  float* scores   = ws + off; off += N;
  int*   idxbuf   = (int*)(ws + off); off += 1024;
  float* vbuf     = ws + off; off += (size_t)C * N;    // v, CN layout
  float* kgbuf    = ws + off; off += (size_t)K * C;    // k_g, row-major
  float* vgbuf    = ws + off; off += (size_t)K * C;    // v_g, row-major
  float* qbuf     = ws + off; off += (size_t)N * C;    // q, row-major
  float* obuf     = ws + off; off += (size_t)N * C;    // attn out, row-major
  float* projbuf  = ws + off; off += (size_t)C * N;    // CN
  float* dwbuf    = ws + off; off += (size_t)C * N;    // CN
  float* pwbuf    = ws + off; off += (size_t)C * N;    // CN
  float* gap      = ws + off; off += 256;
  float* gmp      = ws + off; off += 256;
  float* cabuf    = ws + off; off += 256;
  float* hbuf     = ws + off; off += (size_t)C * N;    // CN
  float* x1buf    = ws + off; off += (size_t)C * N;    // CN
  float* tbuf     = ws + off; off += (size_t)N * 1024; // mlp hidden, row-major
  float* m2buf    = ws + off; off += (size_t)N * C;    // row-major

  float* out0 = (float*)d_out;          // (C, N)
  float* out1 = out0 + (size_t)C * N;   // x2 passthrough

  // 1. spatial attention scores + topk
  meanmax_kernel<<<N / 256, 256, 0, stream>>>(x, spa_mean, spa_max, C, N);
  spa_conv_kernel<<<N / 256, 256, 0, stream>>>(spa_mean, spa_max, sa_w, scores,
                                               D, H, W);
  topk_kernel<<<1, 1024, 0, stream>>>(scores, idxbuf, N, K);

  // 2. projections (bf16 WMMA GEMMs, async-LDS staged)
  // v = tok(x) @ kv_w[C:2C].T + kv_b[C:2C], stored CN
  gemm_tok<true, true, false, 0><<<dim3(N / 64, C / 64), 128, 0, stream>>>(
      x, kv_w + (size_t)C * C, kv_b + C, vbuf, nullptr, N, N, C, C);
  // k_g = gathered tok(x) @ kv_w[:C].T + kv_b[:C], row-major (512, 256)
  gemm_tok<true, false, true, 0><<<dim3(K / 64, C / 64), 128, 0, stream>>>(
      x, kv_w, kv_b, kgbuf, idxbuf, N, K, C, C);
  // v_g gather from v
  gather_vg_kernel<<<(K * C) / 256, 256, 0, stream>>>(vbuf, idxbuf, vgbuf, C, N);
  // q = tok(x2) @ q_w.T + q_b, row-major
  gemm_tok<true, false, false, 0><<<dim3(N / 64, C / 64), 128, 0, stream>>>(
      x2, q_w, q_b, qbuf, nullptr, N, N, C, C);

  // 3. attention (flash-style, WMMA)
  attn_kernel<<<dim3(N / 64, HEADS), 128, 0, stream>>>(qbuf, kgbuf, vgbuf, obuf,
                                                       N, C);
  // proj: o @ proj_w.T + proj_b -> CN
  gemm_tok<false, true, false, 0><<<dim3(N / 64, C / 64), 128, 0, stream>>>(
      obuf, proj_w, proj_b, projbuf, nullptr, N, N, C, C);

  // 4. conv residual path: depthwise then pointwise (WMMA)
  dwconv_kernel<<<(C * N) / 256, 256, 0, stream>>>(vbuf, dw_w, dw_b, dwbuf, C,
                                                   D, H, W);
  gemm_tok<true, true, false, 0><<<dim3(N / 64, C / 64), 128, 0, stream>>>(
      dwbuf, pw_w, pw_b, pwbuf, nullptr, N, N, C, C);

  // 5. channel attention
  gap_gmp_kernel<<<C, 256, 0, stream>>>(x, gap, gmp, N);
  ca_fc_kernel<<<1, 256, 0, stream>>>(gap, gmp, ca_w1, ca_w2, cabuf, C, 16);

  // 6. h = x*ca + proj + pw ; x1 = x + LN1(h)
  h_kernel<<<(C * N) / 256, 256, 0, stream>>>(x, cabuf, projbuf, pwbuf, hbuf, C,
                                              N);
  ln1_kernel<<<N / 256, 256, 0, stream>>>(x, hbuf, ln1_g, ln1_b, x1buf, C, N);

  // 7. MLP (WMMA): t = gelu(x1 @ w1.T + b1) @ w2.T + b2
  gemm_tok<true, false, false, 1><<<dim3(N / 64, 1024 / 64), 128, 0, stream>>>(
      x1buf, mlp_w1, mlp_b1, tbuf, nullptr, N, N, C, 1024);
  gemm_tok<false, false, false, 0><<<dim3(N / 64, C / 64), 128, 0, stream>>>(
      tbuf, mlp_w2, mlp_b2, m2buf, nullptr, N, N, 1024, C);

  // 8. out = x1 + LN2(t)
  ln2_kernel<<<N / 256, 256, 0, stream>>>(x1buf, m2buf, ln2_g, ln2_b, out0, C,
                                          N);

  // 9. second tuple element: x2 passthrough
  hipMemcpyAsync(out1, x2, sizeof(float) * (size_t)C * N,
                 hipMemcpyDeviceToDevice, stream);
}
